// AdaptiveCfCCell_51531017617697
// MI455X (gfx1250) — matrix operand order repaired and checked by reading, not compiled, not run on hardware
//
#include <hip/hip_runtime.h>
#include <hip/hip_bf16.h>
#include <stdint.h>

// ---------------------------------------------------------------------------
// AdaptiveCfCCell forward for MI455X (gfx1250, wave32, WMMA bf16 16x16x32)
//
//   B=8192, D_IN=4096, D_H=512
//   pre1   = [x|h] @ W1 + b1                       (bf16 WMMA, fp32 acc)
//   t      = gelu(layernorm(pre1))                 (fused, stored bf16)
//   params = t @ W2 + b2                           (bf16 WMMA)
//   rec    = h @ base_A^T ; inp = x @ base_B       (bf16 WMMA)
//   out    = layernorm(h + 0.1*(-h/tau + tanh(rec*Ar + inp*Bc)))
//
// GEMM tiles are staged global->LDS with the CDNA5 async-DMA path
// (global_load_async_to_lds_b128 + s_wait_asynccnt): no data VGPRs, no
// register spills, copy overlapped with the WMMA stream.
// ---------------------------------------------------------------------------

typedef __bf16 bf16;
typedef __attribute__((ext_vector_type(16))) __bf16 v16bf;
typedef __attribute__((ext_vector_type(8)))  float  v8f;

union Frag16 {
    uint4 u[2];
    v16bf v;
    bf16  h[16];
};

// ---------------------------------------------------------------------------
// fp32 -> bf16 converts
// ---------------------------------------------------------------------------
__global__ __launch_bounds__(256) void cvt_bf16_kernel(const float* __restrict__ in,
                                                       bf16* __restrict__ out, int n) {
    int i = blockIdx.x * 256 + threadIdx.x;
    if (i < n) out[i] = (bf16)in[i];
}

// in: fp32 [K][N] row-major  ->  out: bf16 [N][K] row-major (GEMM-ready B^T)
__global__ __launch_bounds__(256) void cvt_bf16_T_kernel(const float* __restrict__ in,
                                                         bf16* __restrict__ out,
                                                         int K, int N) {
    int i = blockIdx.x * 256 + threadIdx.x;
    if (i < K * N) {
        int n = i / K, k = i - n * K;
        out[i] = (bf16)in[(size_t)k * N + n];
    }
}

// ---------------------------------------------------------------------------
// Generic bf16 GEMM:  C[M,N](fp32) = concat(A0[M,K0], A1[M,K1]) @ Bt^T (+bias)
//   Bt is the weight stored transposed: [N][K] bf16 row-major.
//   - 256 threads = 8 waves, workgroup tile 128(M) x 128(N), K-step 32
//   - waves arranged 4(M) x 2(N); wave tile 32x64 -> 2x4 WMMA C fragments
//   - double-buffered LDS (rows padded to 40 halves to spread banks)
//   - staging via global_load_async_to_lds_b128 (ASYNCcnt), plus
//     global_prefetch_b8 of the tile after next
//   - requires: M%128==0, N%128==0, K%32==0, K0%32==0
// ---------------------------------------------------------------------------
#define LDS_STRIDE 40   // 32 data halves + 8 pad; 80B row stride, 16B aligned

__global__ __launch_bounds__(256)
void gemm_bf16_kernel(const bf16* __restrict__ A0, int K0,
                      const bf16* __restrict__ A1, int K1,
                      const bf16* __restrict__ Bt,   // [N][K]
                      const float* __restrict__ bias,
                      float* __restrict__ C,
                      int M, int N, int K) {
    __shared__ bf16 As[2][128 * LDS_STRIDE];   // [m_local][k_local]
    __shared__ bf16 Bs[2][128 * LDS_STRIDE];   // [n_local][k_local]

    const int m0 = blockIdx.y * 128;
    const int n0 = blockIdx.x * 128;

    const int tid    = threadIdx.x;
    const int lane   = tid & 31;
    const int wid    = tid >> 5;
    const int wm     = wid & 3;        // 0..3 -> M offset wm*32
    const int wn     = wid >> 2;       // 0..1 -> N offset wn*64
    const int lane16 = lane & 15;
    const int sel    = lane >> 4;      // 0: lanes 0-15, 1: lanes 16-31

    // --- staging index maps: one 32B run per thread per matrix --------------
    const int srow = tid >> 1;              // 0..127 (row in A and B tiles)
    const int sk   = (tid & 1) * 16;        // 0 or 16 (k-halves within row)

    auto aPtr = [&](int kt) -> const bf16* {
        int kg = kt * 32 + sk;
        if (kg < K0) return A0 + (size_t)(m0 + srow) * K0 + kg;
        return A1 + (size_t)(m0 + srow) * K1 + (kg - K0);
    };
    auto bPtr = [&](int kt) -> const bf16* {
        return Bt + (size_t)(n0 + srow) * K + kt * 32 + sk;
    };

    // async DMA: 2x b128 per matrix per thread, directly into LDS.
    // INST_OFFSET is added to BOTH the LDS and global address (ISA 10.7),
    // so one addr pair covers the 32B run.
    auto issueAsync = [&](int kt, int buf) {
        const bf16* pa = aPtr(kt);
        const bf16* pb = bPtr(kt);
        unsigned la = (unsigned)(uintptr_t)&As[buf][srow * LDS_STRIDE + sk];
        unsigned lb = (unsigned)(uintptr_t)&Bs[buf][srow * LDS_STRIDE + sk];
        asm volatile(
            "global_load_async_to_lds_b128 %0, %2, off\n\t"
            "global_load_async_to_lds_b128 %0, %2, off offset:16\n\t"
            "global_load_async_to_lds_b128 %1, %3, off\n\t"
            "global_load_async_to_lds_b128 %1, %3, off offset:16"
            :: "v"(la), "v"(lb), "v"(pa), "v"(pb)
            : "memory");
    };

    v8f acc[2][4] = {};

    issueAsync(0, 0);
    asm volatile("s_wait_asynccnt 0x0" ::: "memory");
    __syncthreads();

    const int T = K >> 5;
    for (int kt = 0; kt < T; ++kt) {
        const int cur = kt & 1;

        if (kt + 1 < T) issueAsync(kt + 1, (kt + 1) & 1);
        if (kt + 2 < T) {                       // global_prefetch_b8 tile after next
            __builtin_prefetch(aPtr(kt + 2), 0, 0);
            __builtin_prefetch(bPtr(kt + 2), 0, 0);
        }

        // --- A fragments: 16x32 bf16, ISA 7.12.2 layout ---------------------
        //   lanes 0-15 (M=lane):     elems 0-7 -> K 0-7,  elems 8-15 -> K 16-23
        //   lanes 16-31 (M=lane-16): elems 0-7 -> K 8-15, elems 8-15 -> K 24-31
        v16bf af[2];
#pragma unroll
        for (int mf = 0; mf < 2; ++mf) {
            const bf16* p = &As[cur][(wm * 32 + mf * 16 + lane16) * LDS_STRIDE];
            Frag16 f;
            f.u[0] = *(const uint4*)(p + sel * 8);
            f.u[1] = *(const uint4*)(p + 16 + sel * 8);
            af[mf] = f.v;
        }
        // --- B fragments: 32x16 bf16 (same K order as A: elems 0-7 -> K 8*sel.., elems 8-15 -> K 16+8*sel..)
        v16bf bfr[4];
#pragma unroll
        for (int nf = 0; nf < 4; ++nf) {
            const bf16* p = &Bs[cur][(wn * 64 + nf * 16 + lane16) * LDS_STRIDE];
            Frag16 f;
            f.u[0] = *(const uint4*)(p + sel * 8);
            f.u[1] = *(const uint4*)(p + 16 + sel * 8);
            bfr[nf] = f.v;
        }

#pragma unroll
        for (int mf = 0; mf < 2; ++mf)
#pragma unroll
            for (int nf = 0; nf < 4; ++nf)
                acc[mf][nf] = __builtin_amdgcn_wmma_f32_16x16x32_bf16(
                    false, af[mf], false, bfr[nf], (short)0, acc[mf][nf],
                    false, false);

        if (kt + 1 < T) asm volatile("s_wait_asynccnt 0x0" ::: "memory");
        __syncthreads();
    }

    // --- writeback: C/D layout = 8 VGPRs, lanes 0-15 M=r, lanes 16-31 M=8+r -
    const int m_base = m0 + wm * 32;
    const int n_base = n0 + wn * 64;
#pragma unroll
    for (int mf = 0; mf < 2; ++mf) {
#pragma unroll
        for (int nf = 0; nf < 4; ++nf) {
            const int col = n_base + nf * 16 + lane16;
            const float bv = bias ? bias[col] : 0.0f;
#pragma unroll
            for (int r = 0; r < 8; ++r) {
                const int row = m_base + mf * 16 + sel * 8 + r;
                C[(size_t)row * N + col] = acc[mf][nf][r] + bv;
            }
        }
    }
}

// ---------------------------------------------------------------------------
// row-wise reduce helper (256 threads, 2 values each, row length 512)
// ---------------------------------------------------------------------------
__device__ inline void row_meanvar_512(float v0, float v1, float& mu, float& inv) {
    float s = v0 + v1, s2 = v0 * v0 + v1 * v1;
#pragma unroll
    for (int off = 16; off > 0; off >>= 1) {
        s  += __shfl_xor(s,  off, 32);
        s2 += __shfl_xor(s2, off, 32);
    }
    __shared__ float rs[8], rs2[8];
    const int lane = threadIdx.x & 31, wv = threadIdx.x >> 5;
    if (lane == 0) { rs[wv] = s; rs2[wv] = s2; }
    __syncthreads();
    float S = 0.f, S2 = 0.f;
#pragma unroll
    for (int i = 0; i < 8; ++i) { S += rs[i]; S2 += rs2[i]; }
    mu = S * (1.0f / 512.0f);
    const float var = S2 * (1.0f / 512.0f) - mu * mu;
    inv = rsqrtf(var + 1e-5f);
}

// t = bf16(gelu_exact(layernorm(pre1; g,b)))    one block per row
__global__ __launch_bounds__(256)
void ln_gelu_kernel(const float* __restrict__ pre, const float* __restrict__ g,
                    const float* __restrict__ b, bf16* __restrict__ t) {
    const size_t row = blockIdx.x;
    const float* p = pre + row * 512;
    const int j0 = threadIdx.x, j1 = threadIdx.x + 256;
    const float v0 = p[j0], v1 = p[j1];
    float mu, inv;
    row_meanvar_512(v0, v1, mu, inv);
    const float y0 = (v0 - mu) * inv * g[j0] + b[j0];
    const float y1 = (v1 - mu) * inv * g[j1] + b[j1];
    const float k = 0.70710678118654752f;      // 1/sqrt(2)  (exact GELU)
    t[row * 512 + j0] = (bf16)(0.5f * y0 * (1.0f + erff(y0 * k)));
    t[row * 512 + j1] = (bf16)(0.5f * y1 * (1.0f + erff(y1 * k)));
}

// final fused CfC update + layernorm      one block per row
__global__ __launch_bounds__(256)
void cfc_final_kernel(const float* __restrict__ h, const float* __restrict__ params,
                      const float* __restrict__ rec, const float* __restrict__ inp,
                      const float* __restrict__ base_tau,
                      const float* __restrict__ hn_g, const float* __restrict__ hn_b,
                      float* __restrict__ out) {
    const size_t row = blockIdx.x;
    const float* pr = params + row * 1536;
    const float* hr = h   + row * 512;
    const float* rr = rec + row * 512;
    const float* ir = inp + row * 512;

    float pre[2];
#pragma unroll
    for (int e = 0; e < 2; ++e) {
        const int j = threadIdx.x + e * 256;
        const float tau_s = pr[j], A_s = pr[512 + j], B_s = pr[1024 + j];
        const float tau   = base_tau[j] * (2.0f / (1.0f + expf(-tau_s)));
        const float A_row = 1.0f + 0.5f * tanhf(A_s);
        const float B_col = 1.0f + 0.5f * tanhf(B_s);
        const float hv    = hr[j];
        const float dhdt  = -hv / (tau + 1e-8f) + tanhf(rr[j] * A_row + ir[j] * B_col);
        pre[e] = hv + 0.1f * dhdt;
    }
    float mu, inv;
    row_meanvar_512(pre[0], pre[1], mu, inv);
#pragma unroll
    for (int e = 0; e < 2; ++e) {
        const int j = threadIdx.x + e * 256;
        out[row * 512 + j] = (pre[e] - mu) * inv * hn_g[j] + hn_b[j];
    }
}

// ---------------------------------------------------------------------------
// launch
// ---------------------------------------------------------------------------
extern "C" void kernel_launch(void* const* d_in, const int* in_sizes, int n_in,
                              void* d_out, int out_size, void* d_ws, size_t ws_size,
                              hipStream_t stream) {
    (void)in_sizes; (void)n_in; (void)out_size; (void)ws_size;
    constexpr int B = 8192, DIN = 4096, DH = 512;

    const float* x    = (const float*)d_in[0];
    const float* h    = (const float*)d_in[1];
    const float* W1   = (const float*)d_in[2];
    const float* b1   = (const float*)d_in[3];
    const float* ln1g = (const float*)d_in[4];
    const float* ln1b = (const float*)d_in[5];
    const float* W2   = (const float*)d_in[6];
    const float* b2   = (const float*)d_in[7];
    const float* btau = (const float*)d_in[8];
    const float* bA   = (const float*)d_in[9];
    const float* bB   = (const float*)d_in[10];
    const float* hng  = (const float*)d_in[11];
    const float* hnb  = (const float*)d_in[12];
    float* out = (float*)d_out;

    // workspace carve-up (256B aligned)
    char* w = (char*)d_ws;
    auto carve = [&](size_t bytes) -> char* {
        char* p = w; w += (bytes + 255) & ~(size_t)255; return p;
    };
    bf16*  xb     = (bf16*)carve((size_t)B * DIN * 2);       // 64 MB
    bf16*  hb     = (bf16*)carve((size_t)B * DH * 2);        //  8 MB
    bf16*  W1t    = (bf16*)carve((size_t)(DIN + DH) * DH * 2);   // [512][4608]
    bf16*  W2t    = (bf16*)carve((size_t)DH * 3 * DH * 2);       // [1536][512]
    bf16*  At     = (bf16*)carve((size_t)DH * DH * 2);           // base_A (== (A^T)^T)
    bf16*  Bbt    = (bf16*)carve((size_t)DIN * DH * 2);          // [512][4096]
    float* pre1   = (float*)carve((size_t)B * DH * 4);       // 16 MB
    bf16*  tb     = (bf16*)carve((size_t)B * DH * 2);        //  8 MB
    float* recw   = (float*)carve((size_t)B * DH * 4);       // 16 MB
    float* inpw   = (float*)carve((size_t)B * DH * 4);       // 16 MB
    float* paramw = (float*)carve((size_t)B * 3 * DH * 4);   // 48 MB

    auto blocks = [](int n) { return (n + 255) / 256; };

    // 1) fp32 -> bf16 activations; weights converted AND transposed to [N][K]
    cvt_bf16_kernel<<<blocks(B * DIN), 256, 0, stream>>>(x, xb, B * DIN);
    cvt_bf16_kernel<<<blocks(B * DH), 256, 0, stream>>>(h, hb, B * DH);
    cvt_bf16_T_kernel<<<blocks((DIN + DH) * DH), 256, 0, stream>>>(W1, W1t, DIN + DH, DH);
    cvt_bf16_T_kernel<<<blocks(DH * 3 * DH), 256, 0, stream>>>(W2, W2t, DH, 3 * DH);
    cvt_bf16_T_kernel<<<blocks(DIN * DH), 256, 0, stream>>>(bB, Bbt, DIN, DH);
    // rec GEMM's transposed weight (base_A^T)^T is base_A itself: plain convert
    cvt_bf16_kernel<<<blocks(DH * DH), 256, 0, stream>>>(bA, At, DH * DH);

    // 2) WMMA GEMMs
    dim3 gridN512(DH / 128, B / 128);         // (4, 64)
    // pre1 = [x|h] @ W1 + b1          (M=8192, N=512, K=4608)
    gemm_bf16_kernel<<<gridN512, 256, 0, stream>>>(xb, DIN, hb, DH, W1t, b1, pre1,
                                                   B, DH, DIN + DH);
    // inp = x @ base_B                (K=4096)
    gemm_bf16_kernel<<<gridN512, 256, 0, stream>>>(xb, DIN, nullptr, 0, Bbt, nullptr,
                                                   inpw, B, DH, DIN);
    // rec = h @ base_A^T              (K=512)
    gemm_bf16_kernel<<<gridN512, 256, 0, stream>>>(hb, DH, nullptr, 0, At, nullptr,
                                                   recw, B, DH, DH);

    // 3) t = bf16(gelu(layernorm(pre1)))
    ln_gelu_kernel<<<B, 256, 0, stream>>>(pre1, ln1g, ln1b, tb);

    // 4) params = t @ W2 + b2         (N=1536, K=512)
    dim3 gridN1536(3 * DH / 128, B / 128);    // (12, 64)
    gemm_bf16_kernel<<<gridN1536, 256, 0, stream>>>(tb, DH, nullptr, 0, W2t, b2,
                                                    paramw, B, 3 * DH, DH);

    // 5) fused CfC update + final layernorm
    cfc_final_kernel<<<B, 256, 0, stream>>>(h, paramw, recw, inpw, btau, hng, hnb, out);
}
